// InteractionDiscovery_28260884807823
// MI455X (gfx1250) — compile-verified, matrix operand order checked by reading, not run on hardware
//
#include <hip/hip_runtime.h>

typedef unsigned int u32;
typedef u32   v8u  __attribute__((ext_vector_type(8)));
typedef float v8f  __attribute__((ext_vector_type(8)));
typedef float f32x4 __attribute__((ext_vector_type(4)));
typedef __bf16 v16bf __attribute__((ext_vector_type(16)));

// ---- workspace layout (dword offsets) ----
#define OFF_PI   0
#define OFF_PJ   20
#define OFF_VALS 40
#define OFF_WC1F 64                       // 8*4*32*8   = 8192 dwords
#define OFF_WC2F (OFF_WC1F + 8192)        // 2*2*32*8   = 1024
#define OFF_W2F  (OFF_WC2F + 1024)        // 20*2*2*32*8 = 20480
#define OFF_W1A  (OFF_W2F + 20480)        // 20*32*3*16 = 30720 (W1/b1 in A-frag pair order)
#define OFF_PART (OFF_W1A + 30720)        // (B/128)*20 floats

#define XT_STRIDE 272                     // bf16 elements per LDS x row (pad)
#define HB_STRIDE 72                      // bf16 elements per LDS h row (pad)

__device__ __forceinline__ unsigned short f2bf(float f) {
  u32 u = __builtin_bit_cast(u32, f);
  u32 r = u + 0x7FFFu + ((u >> 16) & 1u);   // round-to-nearest-even
  return (unsigned short)(r >> 16);
}
__device__ __forceinline__ u32 pack2(float a, float b) {
  return (u32)f2bf(a) | ((u32)f2bf(b) << 16);
}
// K index of the first element of the bf16 pair held in VGPR v by this lane
// (16-bit A/B 16x32 fragment layout, ISA 7.12.2)
__device__ __forceinline__ int pairK0(int v, int lane) {
  return 2 * (v & 3) + ((v & 4) ? 16 : 0) + ((lane & 16) ? 8 : 0);
}
__device__ __forceinline__ v8f wmma_bf16(v8u a, v8u b, v8f c) {
  return __builtin_amdgcn_wmma_f32_16x16x32_bf16(
      false, __builtin_bit_cast(v16bf, a),
      false, __builtin_bit_cast(v16bf, b),
      (short)0, c, false, false);
}
// packed bf16 relu-fma chain: relu(xa*w0 + xb*w1 + b) on a bf16 pair (VOP3P)
__device__ __forceinline__ u32 h1_pair(u32 xap, u32 xbp, u32 w0, u32 w1, u32 b, u32 zero) {
  u32 d;
  asm("v_pk_fma_bf16 %0, %1, %2, %3" : "=v"(d) : "v"(xbp), "v"(w1), "v"(b));
  asm("v_pk_fma_bf16 %0, %1, %2, %0" : "+v"(d) : "v"(xap), "v"(w0));
  asm("v_pk_max_num_bf16 %0, %1, %2" : "+v"(d) : "v"(d), "v"(zero));
  return d;
}

// ---------------- kernel 1: top-K pair selection ----------------
__global__ __launch_bounds__(256) void id_topk_kernel(
    const float* __restrict__ w, float* __restrict__ outk, u32* __restrict__ ws) {
  __shared__ float bv[256];
  __shared__ int   bi[256];
  __shared__ u32   used[2048];     // 65536-bit mask over full 256x256 index
  __shared__ int   selI[20];
  __shared__ float selV[20];
  int tid = threadIdx.x;
  for (int i = tid; i < 2048; i += 256) used[i] = 0u;
  __syncthreads();
  for (int round = 0; round < 20; ++round) {
    float best = -3.4e38f; int bidx = 0x7fffffff;
    for (int t = tid; t < 65536; t += 256) {
      int i = t >> 8, j = t & 255;
      if (j <= i) continue;
      if ((used[t >> 5] >> (t & 31)) & 1u) continue;
      float v = w[t];                       // sigmoid is monotonic: compare raw
      if (v > best) { best = v; bidx = t; } // strided ascending => lowest idx on tie
    }
    bv[tid] = best; bi[tid] = bidx;
    __syncthreads();
    for (int s = 128; s > 0; s >>= 1) {
      if (tid < s) {
        float ov = bv[tid + s]; int oi = bi[tid + s];
        if (ov > bv[tid] || (ov == bv[tid] && oi < bi[tid])) { bv[tid] = ov; bi[tid] = oi; }
      }
      __syncthreads();
    }
    if (tid == 0) {
      selI[round] = bi[0];
      selV[round] = 1.0f / (1.0f + __expf(-bv[0]));
      used[bi[0] >> 5] |= (1u << (bi[0] & 31));
    }
    __syncthreads();
  }
  if (tid < 20) {
    int t = selI[tid]; int i = t >> 8, j = t & 255;
    ((int*)ws)[OFF_PI + tid] = i;
    ((int*)ws)[OFF_PJ + tid] = j;
    ((float*)ws)[OFF_VALS + tid] = selV[tid];
    outk[tid] = selV[tid];                 // importances
    outk[41 + 2 * tid] = (float)i;         // selected_pairs
    outk[42 + 2 * tid] = (float)j;
  }
  if (tid == 0) {
    int c = 0;
    for (int k = 0; k < 20; ++k) c += (selV[k] > 0.1f) ? 1 : 0;
    outk[40] = (float)c;                   // active_interactions
  }
}

// ---------------- kernel 2: weight -> bf16 WMMA fragment prep ----------------
__global__ __launch_bounds__(256) void id_prep_kernel(
    const float* __restrict__ Wc1, const float* __restrict__ Wc2,
    const float* __restrict__ W2,  const float* __restrict__ W1,
    const float* __restrict__ b1,  u32* __restrict__ ws) {
  for (int idx = blockIdx.x * blockDim.x + threadIdx.x; idx < 60416;
       idx += gridDim.x * blockDim.x) {
    if (idx < 8192) {                       // Wc1 frags: [kk8][nt4][lane32][v8]
      int d = idx, v = d & 7, lane = (d >> 3) & 31, nt = (d >> 8) & 3, kk = d >> 10;
      int n = nt * 16 + (lane & 15);
      int K = kk * 32 + pairK0(v, lane);
      ws[OFF_WC1F + d] = pack2(Wc1[K * 64 + n], Wc1[(K + 1) * 64 + n]);
    } else if (idx < 9216) {                // Wc2 frags: [kk2][nt2][lane32][v8], N padded to 32
      int d = idx - 8192, v = d & 7, lane = (d >> 3) & 31, nt = (d >> 8) & 1, kk = (d >> 9) & 1;
      int n = nt * 16 + (lane & 15);
      int K = kk * 32 + pairK0(v, lane);
      float a = (n < 20) ? Wc2[K * 20 + n] : 0.0f;
      float b = (n < 20) ? Wc2[(K + 1) * 20 + n] : 0.0f;
      ws[OFF_WC2F + d] = pack2(a, b);
    } else if (idx < 29696) {               // W2 frags: [k20][kk2][nt2][lane32][v8]
      int d = idx - 9216, v = d & 7, lane = (d >> 3) & 31, nt = (d >> 8) & 1, kk = (d >> 9) & 1, k = d >> 10;
      int n = nt * 16 + (lane & 15);
      int K = kk * 32 + pairK0(v, lane);
      ws[OFF_W2F + d] = pack2(W2[(k * 64 + K) * 32 + n], W2[(k * 64 + K + 1) * 32 + n]);
    } else {                                // W1/b1 pairs in h-A-frag order: [k20][lane32][g3][16]
      int d = idx - 29696;
      int v = d & 7, kk = (d >> 3) & 1;
      int g = (d >> 4) % 3;
      int rl = (d >> 4) / 3;
      int lane = rl & 31, k = rl >> 5;
      int h = kk * 32 + pairK0(v, lane);
      u32 p;
      if (g == 0)      p = pack2(W1[(k * 2 + 0) * 64 + h], W1[(k * 2 + 0) * 64 + h + 1]);
      else if (g == 1) p = pack2(W1[(k * 2 + 1) * 64 + h], W1[(k * 2 + 1) * 64 + h + 1]);
      else             p = pack2(b1[k * 64 + h], b1[k * 64 + h + 1]);
      ws[OFF_W1A + d] = p;
    }
  }
}

// ---------------- kernel 3: main fused compute ----------------
__global__ __launch_bounds__(128) void id_main_kernel(
    const float* __restrict__ x,
    const float* __restrict__ bc1, const float* __restrict__ bc2,
    const float* __restrict__ b2,  const float* __restrict__ b3,
    const float* __restrict__ W3,
    u32* __restrict__ ws, float* __restrict__ out, int B) {
  __shared__ unsigned short xt[4][16 * XT_STRIDE];   // bf16 x tile per wave
  __shared__ unsigned short hbuf[4][16 * HB_STRIDE]; // bf16 hidden tile per wave
  __shared__ int  spi[20], spj[20];
  __shared__ float ms0[4][32], ms1[4][32];

  const int tid = threadIdx.x;
  const int w = tid >> 5, l = tid & 31, l16 = l & 15, hi = (l >> 4) & 1;
  if (tid < 20) {
    spi[tid] = ((const int*)ws)[OFF_PI + tid];
    spj[tid] = ((const int*)ws)[OFF_PJ + tid];
  }
  __syncthreads();

  const u32* wc1f = ws + OFF_WC1F;
  const u32* wc2f = ws + OFF_WC2F;
  const u32* w2f  = ws + OFF_W2F;
  const u32* w1a  = ws + OFF_W1A;

  unsigned short* xw = xt[w];
  unsigned short* hb = hbuf[w];
  const u32* xwd = (const u32*)xw;
  const u32* hbd = (const u32*)hb;
  const u32 zero = 0u;

  float mean0 = 0.0f, mean1 = 0.0f;

  for (int t = 0; t < 2; ++t) {
    const int rowbase = blockIdx.x * 128 + w * 32 + t * 16;

    // ---- stage x tile -> LDS bf16 (streamed once: non-temporal) ----
    if (t == 0) __builtin_prefetch(x + (size_t)(rowbase + 16) * 256 + l * 8, 0, 0);
#pragma unroll 4
    for (int r = 0; r < 16; ++r) {
      const f32x4* p = (const f32x4*)(x + (size_t)(rowbase + r) * 256 + l * 8);
      f32x4 f0 = __builtin_nontemporal_load(p);
      f32x4 f1 = __builtin_nontemporal_load(p + 1);
      uint4 d4;
      d4.x = pack2(f0.x, f0.y); d4.y = pack2(f0.z, f0.w);
      d4.z = pack2(f1.x, f1.y); d4.w = pack2(f1.z, f1.w);
      *((uint4*)&xw[r * XT_STRIDE + l * 8]) = d4;
    }

    // ---- GEMM1: Hc = relu(x @ Wc1 + bc1), [16,256]@[256,64] ----
    v8f acc[4];
#pragma unroll
    for (int nt = 0; nt < 4; ++nt) {
      float bb = bc1[nt * 16 + l16];
#pragma unroll
      for (int i = 0; i < 8; ++i) acc[nt][i] = bb;
    }
#pragma unroll
    for (int kk = 0; kk < 8; ++kk) {
      v8u a;
#pragma unroll
      for (int v = 0; v < 8; ++v)
        a[v] = xwd[(l16 * XT_STRIDE + kk * 32 + pairK0(v, l)) >> 1];
#pragma unroll
      for (int nt = 0; nt < 4; ++nt) {
        const uint4* bp = (const uint4*)(wc1f + ((kk * 4 + nt) * 32 + l) * 8);
        uint4 bq0 = bp[0], bq1 = bp[1];
        v8u bfr = {bq0.x, bq0.y, bq0.z, bq0.w, bq1.x, bq1.y, bq1.z, bq1.w};
        acc[nt] = wmma_bf16(a, bfr, acc[nt]);
      }
    }
    // relu -> LDS (bf16, [m][h])
#pragma unroll
    for (int nt = 0; nt < 4; ++nt) {
      int col = nt * 16 + l16;
#pragma unroll
      for (int r = 0; r < 8; ++r)
        hb[(r + hi * 8) * HB_STRIDE + col] = f2bf(fmaxf(acc[nt][r], 0.0f));
    }
    // gather Hc A-fragments (K=64 -> 2 ksteps); used only by GEMM2
    v8u a2[2];
#pragma unroll
    for (int kk = 0; kk < 2; ++kk)
#pragma unroll
      for (int v = 0; v < 8; ++v)
        a2[kk][v] = hbd[(l16 * HB_STRIDE + kk * 32 + pairK0(v, l)) >> 1];

    // ---- GEMM2: ctx = sigmoid(Hc @ Wc2 + bc2), [16,64]@[64,20->32] ----
    v8f cacc[2];
    {
      float bb0 = bc2[l16];
      float bb1 = (l16 < 4) ? bc2[16 + l16] : 0.0f;
#pragma unroll
      for (int i = 0; i < 8; ++i) { cacc[0][i] = bb0; cacc[1][i] = bb1; }
    }
#pragma unroll
    for (int kk = 0; kk < 2; ++kk)
#pragma unroll
      for (int nt = 0; nt < 2; ++nt) {
        const uint4* bp = (const uint4*)(wc2f + ((kk * 2 + nt) * 32 + l) * 8);
        uint4 bq0 = bp[0], bq1 = bp[1];
        v8u bfr = {bq0.x, bq0.y, bq0.z, bq0.w, bq1.x, bq1.y, bq1.z, bq1.w};
        cacc[nt] = wmma_bf16(a2[kk], bfr, cacc[nt]);
      }
    v8f ctx0, ctx1;
#pragma unroll
    for (int r = 0; r < 8; ++r) {
      ctx0[r] = 1.0f / (1.0f + __expf(-cacc[0][r]));
      ctx1[r] = 1.0f / (1.0f + __expf(-cacc[1][r]));
    }
    {
      float s0 = 0.0f, s1 = 0.0f;
#pragma unroll
      for (int r = 0; r < 8; ++r) { s0 += ctx0[r]; s1 += ctx1[r]; }
      s0 += __shfl_xor(s0, 16, 32);
      s1 += __shfl_xor(s1, 16, 32);
      mean0 += s0; mean1 += s1;
    }

    // ---- per-interaction tiny MLPs ----
    for (int k = 0; k < 20; ++k) {
      const int pik = spi[k], pjk = spj[k];
      // broadcast bf16 pair of (xa, xa) and (xb, xb) for this lane's row
      u32 xa = xw[l16 * XT_STRIDE + pik];
      u32 xb = xw[l16 * XT_STRIDE + pjk];
      u32 xap = xa | (xa << 16);
      u32 xbp = xb | (xb << 16);

      // h1 in A-fragment layout directly via packed bf16 VALU (co-executes w/ WMMA)
      const uint4* wp = (const uint4*)(w1a + (size_t)(k * 32 + l) * 48);
      u32 w0p[16], w1p[16], bp_[16];
#pragma unroll
      for (int qi = 0; qi < 4; ++qi) {
        uint4 q0 = wp[qi], q1 = wp[qi + 4], q2 = wp[qi + 8];
        w0p[4 * qi] = q0.x; w0p[4 * qi + 1] = q0.y; w0p[4 * qi + 2] = q0.z; w0p[4 * qi + 3] = q0.w;
        w1p[4 * qi] = q1.x; w1p[4 * qi + 1] = q1.y; w1p[4 * qi + 2] = q1.z; w1p[4 * qi + 3] = q1.w;
        bp_[4 * qi] = q2.x; bp_[4 * qi + 1] = q2.y; bp_[4 * qi + 2] = q2.z; bp_[4 * qi + 3] = q2.w;
      }
      v8u a3[2];
#pragma unroll
      for (int e = 0; e < 16; ++e)
        a3[e >> 3][e & 7] = h1_pair(xap, xbp, w0p[e], w1p[e], bp_[e], zero);

      // h2 = relu(h1 @ W2[k] + b2[k]), [16,64]@[64,32]
      v8f h2acc[2];
#pragma unroll
      for (int nt = 0; nt < 2; ++nt) {
        float bb = b2[k * 32 + nt * 16 + l16];
#pragma unroll
        for (int i = 0; i < 8; ++i) h2acc[nt][i] = bb;
      }
#pragma unroll
      for (int kk = 0; kk < 2; ++kk)
#pragma unroll
        for (int nt = 0; nt < 2; ++nt) {
          const uint4* bp = (const uint4*)(w2f + (((k * 2 + kk) * 2 + nt) * 32 + l) * 8);
          uint4 bq0 = bp[0], bq1 = bp[1];
          v8u bfr = {bq0.x, bq0.y, bq0.z, bq0.w, bq1.x, bq1.y, bq1.z, bq1.w};
          h2acc[nt] = wmma_bf16(a3[kk], bfr, h2acc[nt]);
        }

      // out[m,k] = relu(h2) . W3[k] + b3[k]; butterfly-reduce over 16 lanes
      float s[8];
#pragma unroll
      for (int r = 0; r < 8; ++r) s[r] = 0.0f;
#pragma unroll
      for (int nt = 0; nt < 2; ++nt) {
        float w3c = W3[k * 32 + nt * 16 + l16];
#pragma unroll
        for (int r = 0; r < 8; ++r) s[r] += fmaxf(h2acc[nt][r], 0.0f) * w3c;
      }
#pragma unroll
      for (int off = 1; off < 16; off <<= 1)
#pragma unroll
        for (int r = 0; r < 8; ++r) s[r] += __shfl_xor(s[r], off, 32);

      float b3k = b3[k];
      int src = ((k < 16) ? k : (k - 16)) + hi * 16;
      float cv[8];
#pragma unroll
      for (int r = 0; r < 8; ++r)
        cv[r] = (k < 16) ? __shfl(ctx0[r], src, 32) : __shfl(ctx1[r], src, 32);

      if (l16 == 0) {
#pragma unroll
        for (int r = 0; r < 8; ++r)
          __builtin_nontemporal_store(
              (s[r] + b3k) * cv[r],
              out + (size_t)(rowbase + hi * 8 + r) * 20 + k);
      }
    }
  }

  // block-level deterministic ctx-mean partials
  ms0[w][l] = mean0;
  ms1[w][l] = mean1;
  __syncthreads();
  if (tid < 20) {
    float s = 0.0f;
    for (int wv = 0; wv < 4; ++wv)
      s += (tid < 16) ? ms0[wv][tid] : ms1[wv][tid - 16];
    ((float*)(ws + OFF_PART))[blockIdx.x * 20 + tid] = s;
  }
  (void)B;
}

// ---------------- kernel 4: fixed-order mean reduction ----------------
__global__ void id_finalize_kernel(const u32* __restrict__ ws,
                                   float* __restrict__ outMean, int B) {
  int k = threadIdx.x;
  if (k < 20) {
    int nb = B / 128;
    const float* part = (const float*)(ws + OFF_PART);
    float s = 0.0f;
    for (int b = 0; b < nb; ++b) s += part[b * 20 + k];
    outMean[k] = s / (float)B;
  }
}

extern "C" void kernel_launch(void* const* d_in, const int* in_sizes, int n_in,
                              void* d_out, int out_size, void* d_ws, size_t ws_size,
                              hipStream_t stream) {
  const float* x   = (const float*)d_in[0];
  const float* iw  = (const float*)d_in[1];
  const float* Wc1 = (const float*)d_in[2];
  const float* bc1 = (const float*)d_in[3];
  const float* Wc2 = (const float*)d_in[4];
  const float* bc2 = (const float*)d_in[5];
  const float* W1  = (const float*)d_in[6];
  const float* b1  = (const float*)d_in[7];
  const float* W2  = (const float*)d_in[8];
  const float* b2  = (const float*)d_in[9];
  const float* W3  = (const float*)d_in[10];
  const float* b3  = (const float*)d_in[11];
  float* out = (float*)d_out;
  u32*   ws  = (u32*)d_ws;
  const int B = in_sizes[0] / 256;

  id_topk_kernel<<<1, 256, 0, stream>>>(iw, out + (size_t)B * 20, ws);
  id_prep_kernel<<<236, 256, 0, stream>>>(Wc1, Wc2, W2, W1, b1, ws);
  id_main_kernel<<<B / 128, 128, 0, stream>>>(x, bc1, bc2, b2, b3, W3, ws, out, B);
  id_finalize_kernel<<<1, 32, 0, stream>>>(ws, out + (size_t)B * 20 + 20, B);
  (void)n_in; (void)out_size; (void)ws_size;
}
